// Vanilla_SSM_11733850653046
// MI455X (gfx1250) — matrix operand order repaired and checked by reading, not compiled, not run on hardware
//
// MI455X (gfx1250) SSM: y_t = h_t WC^T + x_t WD^T + bC+bD ;  h_{t+1} = h_t WA^T + x_t WB^T + bA+bB
//
// Strategy (latency-bound sequential scan -> stride-S parallel scan):
//   M = WA^T (right-multiplier).  h[t+S] = h[t] M^S + w[t],
//   w[t] = sum_{k=0}^{S-1} u[t+S-1-k] M^k,  u = x WB^T + bU
//        = sum_k x[t+S-1-k] (WA^k WB)^T_conv + bw          (weights precomputed)
//   S=8: 8 independent residue-class scans of 512 steps (8 WGPs in parallel),
//   chain length cut 8x vs naive.  All heavy math = v_wmma_f32_16x16x32_bf16.
//   H (all states) is stored in WMMA A-fragment layout so the final y-GEMM
//   loads fragments from global with coalesced b128, no relayout.
//   Weight-pack staging into LDS uses the Tensor Data Mover
//   (tensor_load_to_lds + s_wait_tensorcnt); B-fragment reads in the GEMM
//   inner loops are software-pipelined (depth 1) so one ds_load_b128 pair is
//   always in flight behind each WMMA.
//
// Roofline: HBM traffic ~0.5 GB -> ~21us @23.3TB/s; bf16 WMMA work ~110 GFLOP
// (parallel); sequential chain 512 steps * ~250 cyc ~ 50us. Balanced design.
//
// Workspace (~101 MB): w[T][16][256] f32 | H frag bf16 T*8KB | WA powers f32 |
// WA^k*WB f32 | 12 packed bf16 B-fragment matrices | bias vectors.

#include <hip/hip_runtime.h>
#include <stdint.h>

#define DI __device__ __forceinline__

typedef float    f32x4 __attribute__((ext_vector_type(4)));
typedef float    v8f   __attribute__((ext_vector_type(8)));
typedef uint32_t u32x4 __attribute__((ext_vector_type(4)));
typedef uint32_t u32x8 __attribute__((ext_vector_type(8)));
typedef __bf16   v16bf __attribute__((ext_vector_type(16)));

constexpr int BB = 16, T = 4096, C = 256, S = 8;
constexpr int KT = 8;              // K tiles of 32 (bf16)
constexpr int NT = 16;             // N tiles of 16
constexpr int FRAG_DW = KT * 32 * 8;   // 2048 dwords: one 16x256 bf16 slice in A-frag layout
constexpr int PACK_DW = KT * NT * 32 * 8; // 32768 dwords: one 256x256 bf16 packed B-frag matrix

// ---- workspace offsets (bytes) ----
constexpr size_t OFF_W  = 0;                                   // w: T*16*256 f32
constexpr size_t OFF_H  = OFF_W + (size_t)T * 16 * 256 * 4;    // H: T slices * FRAG_DW dw
constexpr size_t OFF_P  = OFF_H + (size_t)T * FRAG_DW * 4;     // P[k],k=2..8 : 7 * 256*256 f32
constexpr size_t OFF_Q  = OFF_P + 7ull * 256 * 256 * 4;        // Q[k]=WA^k*WB,k=1..7
constexpr size_t OFF_PK = OFF_Q + 7ull * 256 * 256 * 4;        // 12 packed matrices
constexpr size_t OFF_BU = OFF_PK + 12ull * PACK_DW * 4;
constexpr size_t OFF_BV = OFF_BU + 1024;
constexpr size_t OFF_BW = OFF_BV + 1024;

// ---- bf16 helpers (round to nearest even) ----
DI uint32_t bf16rne(float f) {
  uint32_t u = __float_as_uint(f);
  return (u + 0x7FFFu + ((u >> 16) & 1u)) >> 16;
}
DI uint32_t bfpair(float lo, float hi) {
  return (bf16rne(lo) & 0xFFFFu) | (bf16rne(hi) << 16);
}
DI u32x4 ld4(const uint32_t* p) { return *(const u32x4*)p; }
DI u32x8 cat8(u32x4 a, u32x4 b) {
  u32x8 r; r[0]=a[0]; r[1]=a[1]; r[2]=a[2]; r[3]=a[3];
           r[4]=b[0]; r[5]=b[1]; r[6]=b[2]; r[7]=b[3]; return r;
}
DI v16bf asbf(u32x8 r) { return __builtin_bit_cast(v16bf, r); }
DI v16bf frag_from(const uint32_t* dw) { return asbf(cat8(ld4(dw), ld4(dw + 4))); }
DI v8f wmma_bf16(v16bf a, v16bf b, v8f c) {
  return __builtin_amdgcn_wmma_f32_16x16x32_bf16(false, a, false, b, (short)0, c, false, false);
}

// ---- Tensor Data Mover: contiguous global->LDS copy (ISA ch.7-8 of 08_async_tensor) ----
#if defined(__has_builtin)
#if __has_builtin(__builtin_amdgcn_tensor_load_to_lds)
#define HAVE_TDM 1
#endif
#endif

#ifdef HAVE_TDM
typedef int i32x4 __attribute__((ext_vector_type(4)));
typedef int i32x8 __attribute__((ext_vector_type(8)));

// One wave issues the DMA; EXEC ignored by TDM. nbytes must be a multiple of 8.
// D# group0: count=1 | lds_addr | global_addr[56:0] | type=2.
// D# group1: data_size=8B; tensor_dim0=tile_dim0=nbytes/8 (<=65535); dim1=1.
DI void tdm_copy_to_lds(uint32_t lds_off, const void* gsrc, uint32_t nbytes) {
  uint64_t ga = (uint64_t)(uintptr_t)gsrc;
  uint32_t ne = nbytes >> 3;                                  // 8-byte elements
  u32x4 g0;
  g0[0] = 1u;                                                 // count=1 (valid user D#)
  g0[1] = lds_off;                                            // lds_addr (bytes)
  g0[2] = (uint32_t)(ga & 0xFFFFFFFFu);                       // global_addr[31:0]
  g0[3] = (uint32_t)((ga >> 32) & 0x1FFFFFFu) | (2u << 30);   // global_addr[56:32], type=2
  i32x8 g1 = {0, 0, 0, 0, 0, 0, 0, 0};
  g1[0] = (int)(3u << 16);                                    // data_size = 3 (8 bytes)
  g1[1] = (int)((ne & 0xFFFFu) << 16);                        // tensor_dim0[15:0]
  g1[2] = (int)(((ne >> 16) & 0xFFFFu) | (1u << 16));         // tensor_dim0[31:16], dim1=1
  g1[3] = (int)((ne & 0xFFFFu) << 16);                        // tile_dim0
  g1[4] = 1;                                                  // tile_dim1 = 1
  g1[5] = (int)ne;                                            // tensor_dim0_stride[31:0]
  i32x4 z4 = {0, 0, 0, 0};
#if __clang_major__ >= 23
  i32x8 z8 = {0, 0, 0, 0, 0, 0, 0, 0};
  __builtin_amdgcn_tensor_load_to_lds(g0, g1, z4, z4, z8, 0);
#else
  __builtin_amdgcn_tensor_load_to_lds(g0, g1, z4, z4, 0);
#endif
}
#endif

// A-fragment layout (16x32 bf16 tile, ISA 7.12.2): lane l(0..15)=row l holds
// K%32 in {0..7,16..23}; lane l+16 holds {8..15,24..31}. dword v: pairs.
// Slice layout: frag dword index = (kt*32 + lane)*8 + v  (contiguous 32B/lane).
// B-fragment (32x16): lane = n%16 + 16*h, dword v -> K = kt*32 + (v>>2)*16 + h*8 + 2(v&3){+1}.

// ========================= prep kernels =========================
__global__ void k_gemm256(const float* __restrict__ A, const float* __restrict__ Bm,
                          float* __restrict__ Cm) {  // Cm = A*Bm (256x256 f32)
  int r = blockIdx.x, c = threadIdx.x;
  float acc = 0.f;
  for (int k = 0; k < 256; ++k) acc = fmaf(A[r * 256 + k], Bm[k * 256 + c], acc);
  Cm[r * 256 + c] = acc;
}

__global__ void k_pack(const float* __restrict__ Wt, uint32_t* __restrict__ pb) {
  // pb[((kt*16+nt)*32+lane)*8+v] packs Wt[n][K],Wt[n][K+1]
  int idx = blockIdx.x * 256 + threadIdx.x;      // 128 blocks x 256
  int v = idx & 7, lane = (idx >> 3) & 31, nt = (idx >> 8) & 15, kt = idx >> 12;
  int n = nt * 16 + (lane & 15);
  int K = kt * 32 + (v >> 2) * 16 + (lane >> 4) * 8 + 2 * (v & 3);
  pb[idx] = bfpair(Wt[n * 256 + K], Wt[n * 256 + K + 1]);
}

__global__ void k_bias(const float* bA, const float* bB, const float* bC, const float* bD,
                       const float* __restrict__ Pbase, const float* __restrict__ WA,
                       float* bU, float* bV, float* bw) {
  __shared__ float sbu[256];
  int c = threadIdx.x;
  float u = bA[c] + bB[c];
  bU[c] = u; sbu[c] = u;
  bV[c] = bC[c] + bD[c];
  __syncthreads();
  float acc = u;                                  // k=0 term
  for (int k = 1; k < S; ++k) {                   // + sum_k (WA^k bU)
    const float* M = (k == 1) ? WA : (Pbase + (size_t)(k - 2) * 65536);
    float s = 0.f;
    for (int e = 0; e < 256; ++e) s = fmaf(M[c * 256 + e], sbu[e], s);
    acc += s;
  }
  bw[c] = acc;
}

// ========================= w precompute (68 GFLOP bf16 WMMA) =========================
// block: 8 output t-slices, 8 waves (wave -> one slice, all 16 N-tiles).
// LDS: 15 staged x slices (A-frags) + one packed weight matrix (TDM-staged per k).
__global__ void __launch_bounds__(256) k_wgemm(
    const float* __restrict__ x, const uint32_t* __restrict__ packWw,
    const float* __restrict__ bw, float* __restrict__ wout) {
  extern __shared__ uint32_t sm[];
  uint32_t* xf = sm;                       // 15 * FRAG_DW
  uint32_t* wf = sm + 15 * FRAG_DW;        // PACK_DW (LDS byte offset 15*FRAG_DW*4)
  int tid = threadIdx.x, lane = tid & 31, wid = tid >> 5;
  int t0 = blockIdx.x * 8;

  // stage x slices t0..t0+14 as bf16 A-frags (zero-pad past T)
  for (int i = tid; i < 15 * 1024; i += 256) {
    int sl = i >> 10, j = i & 1023, b = j >> 6, c = (j & 63) * 4;
    int tt = t0 + sl;
    f32x4 p;
    if (tt < T) p = *(const f32x4*)(x + ((size_t)b * T + tt) * C + c);
    else { p[0] = p[1] = p[2] = p[3] = 0.f; }
    int kt = c >> 5, kh = (c >> 4) & 1, r16 = c & 15;
    int dl = b + 16 * (r16 >> 3);
    int v = 4 * kh + ((r16 & 7) >> 1);
    uint32_t* d = &xf[(size_t)sl * FRAG_DW + (kt * 32 + dl) * 8 + v];
    d[0] = bfpair(p[0], p[1]); d[1] = bfpair(p[2], p[3]);
  }

  v8f acc[16];
  for (int j = 0; j < 16; ++j) {
    float bv = bw[j * 16 + (lane & 15)];
    for (int i = 0; i < 8; ++i) acc[j][i] = bv;
  }

  for (int k = 0; k < S; ++k) {
    __syncthreads();                                    // also covers x staging at k=0
#ifdef HAVE_TDM
    if (wid == 0) {                                     // one wave drives the DMA
      tdm_copy_to_lds((uint32_t)(15 * FRAG_DW * 4),
                      packWw + (size_t)k * PACK_DW, (uint32_t)PACK_DW * 4);
      __builtin_amdgcn_s_wait_tensorcnt(0);
    }
#else
    for (int i = tid; i < PACK_DW / 4; i += 256)        // fallback: VGPR round-trip copy
      ((u32x4*)wf)[i] = ((const u32x4*)(packWw + (size_t)k * PACK_DW))[i];
#endif
    __syncthreads();
    int si = wid + 7 - k;                               // source slice for this wave
    for (int kt = 0; kt < KT; ++kt) {
      v16bf a = frag_from(&xf[(size_t)si * FRAG_DW + (kt * 32 + lane) * 8]);
      // depth-1 software pipeline: B-frag for nt+1 in flight behind WMMA for nt
      v16bf bcur = frag_from(&wf[((kt * NT + 0) * 32 + lane) * 8]);
#pragma unroll
      for (int nt = 0; nt < NT - 1; ++nt) {
        v16bf bnxt = frag_from(&wf[((kt * NT + nt + 1) * 32 + lane) * 8]);
        acc[nt] = wmma_bf16(a, bcur, acc[nt]);
        bcur = bnxt;
      }
      acc[NT - 1] = wmma_bf16(a, bcur, acc[NT - 1]);
    }
  }

  int t = t0 + wid;
  int bh = 8 * (lane >> 4), nn = lane & 15;
#pragma unroll
  for (int nt = 0; nt < NT; ++nt)
    for (int rr = 0; rr < 8; ++rr)
      wout[((size_t)t * 16 + rr + bh) * 256 + nt * 16 + nn] = acc[nt][rr];
}

// ========================= bootstrap h_1..h_{S-1} =========================
__global__ void __launch_bounds__(512) k_bootstrap(
    const float* __restrict__ x, const float* __restrict__ h0,
    const uint32_t* __restrict__ packWB, const uint32_t* __restrict__ packWA,
    const float* __restrict__ bU, uint32_t* __restrict__ Hglob) {
  __shared__ uint32_t hf[2][FRAG_DW];
  __shared__ uint32_t xf[FRAG_DW];
  int tid = threadIdx.x, lane = tid & 31, wid = tid >> 5;   // wid = N-tile, 16 waves

  u32x8 fb[KT], fa[KT];                                     // WB / WA B-frags resident
  for (int kt = 0; kt < KT; ++kt) {
    const uint32_t* pB = packWB + ((size_t)(kt * NT + wid) * 32 + lane) * 8;
    const uint32_t* pA = packWA + ((size_t)(kt * NT + wid) * 32 + lane) * 8;
    fb[kt] = cat8(ld4(pB), ld4(pB + 4));
    fa[kt] = cat8(ld4(pA), ld4(pA + 4));
  }
  // h0 -> hf[0] (bf16 A-frags)
  {
    int b = tid >> 5, c8 = (tid & 31) * 8;
    f32x4 p0 = *(const f32x4*)(h0 + b * 256 + c8);
    f32x4 p1 = *(const f32x4*)(h0 + b * 256 + c8 + 4);
    int kt = c8 >> 5, kh = (c8 >> 4) & 1, r16 = c8 & 15;
    int dl = b + 16 * (r16 >> 3);
    uint32_t* d = &hf[0][(kt * 32 + dl) * 8 + 4 * kh];
    d[0] = bfpair(p0[0], p0[1]); d[1] = bfpair(p0[2], p0[3]);
    d[2] = bfpair(p1[0], p1[1]); d[3] = bfpair(p1[2], p1[3]);
  }
  __syncthreads();
  ((u32x4*)Hglob)[tid] = ((const u32x4*)hf[0])[tid];        // H[0]
  float bu = bU[wid * 16 + (lane & 15)];
  int cur = 0;
  for (int t = 0; t < S - 1; ++t) {
    {  // stage x_t
      int b = tid >> 5, c8 = (tid & 31) * 8;
      const float* xp = x + ((size_t)b * T + t) * C + c8;
      f32x4 p0 = *(const f32x4*)xp, p1 = *(const f32x4*)(xp + 4);
      int kt = c8 >> 5, kh = (c8 >> 4) & 1, r16 = c8 & 15;
      int dl = b + 16 * (r16 >> 3);
      uint32_t* d = &xf[(kt * 32 + dl) * 8 + 4 * kh];
      d[0] = bfpair(p0[0], p0[1]); d[1] = bfpair(p0[2], p0[3]);
      d[2] = bfpair(p1[0], p1[1]); d[3] = bfpair(p1[2], p1[3]);
    }
    __syncthreads();
    v8f acc; for (int i = 0; i < 8; ++i) acc[i] = bu;
#pragma unroll
    for (int kt = 0; kt < KT; ++kt)
      acc = wmma_bf16(frag_from(&xf[(kt * 32 + lane) * 8]), asbf(fb[kt]), acc);
#pragma unroll
    for (int kt = 0; kt < KT; ++kt)
      acc = wmma_bf16(frag_from(&hf[cur][(kt * 32 + lane) * 8]), asbf(fa[kt]), acc);
    int nxt = cur ^ 1;
    {  // D-tile -> A-frag scatter (transpose via LDS b16 stores)
      int kt = wid >> 1, kh = wid & 1, nn = lane & 15;
      int v = 4 * kh + ((nn & 7) >> 1), bo = nn & 1, dl0 = 16 * (nn >> 3);
      unsigned short* hp = (unsigned short*)hf[nxt];
      for (int rr = 0; rr < 8; ++rr) {
        int b = rr + 8 * (lane >> 4);
        hp[((((kt * 32 + b + dl0) * 8) + v) << 1) + bo] = (unsigned short)bf16rne(acc[rr]);
      }
    }
    __syncthreads();
    ((u32x4*)(Hglob + (size_t)(t + 1) * FRAG_DW))[tid] = ((const u32x4*)hf[nxt])[tid];
    cur = nxt;
  }
}

// ========================= stride-8 parallel scan =========================
// 8 blocks (residue classes) x 16 waves. WA^8 B-frags resident in VGPRs;
// double-buffered bf16 state in LDS; one barrier/step; w[t+8] prefetched.
__global__ void __launch_bounds__(512) k_sscan(
    const float* __restrict__ w, const uint32_t* __restrict__ packP8,
    uint32_t* __restrict__ Hglob, float* __restrict__ hlast) {
  __shared__ uint32_t hf[2][FRAG_DW];
  int r = blockIdx.x;
  int tid = threadIdx.x, lane = tid & 31, wid = tid >> 5;
  u32x8 fr[KT];
  for (int kt = 0; kt < KT; ++kt) {
    const uint32_t* p = packP8 + ((size_t)(kt * NT + wid) * 32 + lane) * 8;
    fr[kt] = cat8(ld4(p), ld4(p + 4));
  }
  ((u32x4*)hf[0])[tid] = ((const u32x4*)(Hglob + (size_t)r * FRAG_DW))[tid];  // state = h_r
  __syncthreads();

  int nfull = (T - r) / S;
  int bh = 8 * (lane >> 4), nn = lane & 15;
  int nidx = wid * 16 + nn;
  v8f wc;
  for (int rr = 0; rr < 8; ++rr) wc[rr] = w[((size_t)r * 16 + rr + bh) * 256 + nidx];
  int cur = 0;
  for (int it = 0; it < nfull; ++it) {
    int t = r + it * S, tn = t + S;
    v8f wn;                                  // prefetch next step's w
    int tpre = (it + 1 < nfull) ? tn : t;
    for (int rr = 0; rr < 8; ++rr) wn[rr] = w[((size_t)tpre * 16 + rr + bh) * 256 + nidx];
    v8f acc = wc;
#pragma unroll
    for (int kt = 0; kt < KT; ++kt)
      acc = wmma_bf16(frag_from(&hf[cur][(kt * 32 + lane) * 8]), asbf(fr[kt]), acc);
    int nxt = cur ^ 1;
    {
      int kt = wid >> 1, kh = wid & 1;
      int v = 4 * kh + ((nn & 7) >> 1), bo = nn & 1, dl0 = 16 * (nn >> 3);
      unsigned short* hp = (unsigned short*)hf[nxt];
      for (int rr = 0; rr < 8; ++rr) {
        int b = rr + bh;
        hp[((((kt * 32 + b + dl0) * 8) + v) << 1) + bo] = (unsigned short)bf16rne(acc[rr]);
      }
    }
    __syncthreads();
    if (tn < T) {
      ((u32x4*)(Hglob + (size_t)tn * FRAG_DW))[tid] = ((const u32x4*)hf[nxt])[tid];
    } else {                                  // class 0, tn==T: h_last (fp32)
      for (int rr = 0; rr < 8; ++rr) hlast[(rr + bh) * 256 + nidx] = acc[rr];
    }
    wc = wn; cur = nxt;
  }
}

// ========================= y = H*WC^T + x*WD^T + bV (34 GFLOP) =========================
__global__ void __launch_bounds__(256) k_ygemm(
    const float* __restrict__ x, const uint32_t* __restrict__ Hglob,
    const uint32_t* __restrict__ packWC, const uint32_t* __restrict__ packWD,
    const float* __restrict__ bV, float* __restrict__ y) {
  extern __shared__ uint32_t sm[];
  uint32_t* xf = sm;                     // 4 * FRAG_DW
  uint32_t* wc = sm + 4 * FRAG_DW;       // PACK_DW   (LDS byte offset 32768)
  uint32_t* wd = wc + PACK_DW;           // PACK_DW   (LDS byte offset 163840)
  int tid = threadIdx.x, lane = tid & 31, wid = tid >> 5;
  int t0 = blockIdx.x * 4;

#ifdef HAVE_TDM
  if (wid == 0) {                        // DMA both weight packs while x is staged
    tdm_copy_to_lds((uint32_t)(4 * FRAG_DW * 4), packWC, (uint32_t)PACK_DW * 4);
    tdm_copy_to_lds((uint32_t)(4 * FRAG_DW * 4 + PACK_DW * 4), packWD,
                    (uint32_t)PACK_DW * 4);
  }
#endif
  for (int i = tid; i < 4 * 1024; i += 256) {       // stage 4 x slices
    int sl = i >> 10, j = i & 1023, b = j >> 6, c = (j & 63) * 4;
    f32x4 p = *(const f32x4*)(x + ((size_t)b * T + (t0 + sl)) * C + c);
    int kt = c >> 5, kh = (c >> 4) & 1, r16 = c & 15;
    int dl = b + 16 * (r16 >> 3);
    int v = 4 * kh + ((r16 & 7) >> 1);
    uint32_t* d = &xf[(size_t)sl * FRAG_DW + (kt * 32 + dl) * 8 + v];
    d[0] = bfpair(p[0], p[1]); d[1] = bfpair(p[2], p[3]);
  }
#ifdef HAVE_TDM
  if (wid == 0) __builtin_amdgcn_s_wait_tensorcnt(0);
#else
  for (int i = tid; i < PACK_DW / 4; i += 256) ((u32x4*)wc)[i] = ((const u32x4*)packWC)[i];
  for (int i = tid; i < PACK_DW / 4; i += 256) ((u32x4*)wd)[i] = ((const u32x4*)packWD)[i];
#endif
  __syncthreads();

  int mt = wid & 3, nh = wid >> 2;
  int t = t0 + mt;
  int nn = lane & 15, bh = 8 * (lane >> 4);
  v8f acc[8];
  for (int j = 0; j < 8; ++j) {
    float bv = bV[(nh * 8 + j) * 16 + nn];
    for (int i = 0; i < 8; ++i) acc[j][i] = bv;
  }
  const uint32_t* Hs = Hglob + (size_t)t * FRAG_DW;  // H already in A-frag layout
  for (int kt = 0; kt < KT; ++kt) {
    v16bf aH = frag_from(&Hs[(kt * 32 + lane) * 8]);   // coalesced global b128 x2
    v16bf bcur = frag_from(&wc[((kt * NT + nh * 8) * 32 + lane) * 8]);
#pragma unroll
    for (int j = 0; j < 7; ++j) {
      v16bf bnxt = frag_from(&wc[((kt * NT + nh * 8 + j + 1) * 32 + lane) * 8]);
      acc[j] = wmma_bf16(aH, bcur, acc[j]);
      bcur = bnxt;
    }
    acc[7] = wmma_bf16(aH, bcur, acc[7]);
  }
  for (int kt = 0; kt < KT; ++kt) {
    v16bf aX = frag_from(&xf[(size_t)mt * FRAG_DW + (kt * 32 + lane) * 8]);
    v16bf bcur = frag_from(&wd[((kt * NT + nh * 8) * 32 + lane) * 8]);
#pragma unroll
    for (int j = 0; j < 7; ++j) {
      v16bf bnxt = frag_from(&wd[((kt * NT + nh * 8 + j + 1) * 32 + lane) * 8]);
      acc[j] = wmma_bf16(aX, bcur, acc[j]);
      bcur = bnxt;
    }
    acc[7] = wmma_bf16(aX, bcur, acc[7]);
  }
#pragma unroll
  for (int j = 0; j < 8; ++j) {
    int n = (nh * 8 + j) * 16 + nn;
    for (int rr = 0; rr < 8; ++rr)
      y[((size_t)(rr + bh) * T + t) * C + n] = acc[j][rr];
  }
}

// ========================= host side =========================
extern "C" void kernel_launch(void* const* d_in, const int* in_sizes, int n_in,
                              void* d_out, int out_size, void* d_ws, size_t ws_size,
                              hipStream_t stream) {
  (void)in_sizes; (void)n_in; (void)out_size; (void)ws_size;
  const float* x  = (const float*)d_in[0];
  const float* h0 = (const float*)d_in[1];
  const float* WA = (const float*)d_in[2];
  const float* bA = (const float*)d_in[3];
  const float* WB = (const float*)d_in[4];
  const float* bB = (const float*)d_in[5];
  const float* WC = (const float*)d_in[6];
  const float* bC = (const float*)d_in[7];
  const float* WD = (const float*)d_in[8];
  const float* bD = (const float*)d_in[9];
  float* out_y = (float*)d_out;
  float* out_h = out_y + (size_t)BB * T * C;

  uint8_t* ws = (uint8_t*)d_ws;
  float*    wbuf = (float*)(ws + OFF_W);
  uint32_t* H    = (uint32_t*)(ws + OFF_H);
  float*    P    = (float*)(ws + OFF_P);    // P[k]=WA^k, k=2..8 at (k-2)*65536
  float*    Q    = (float*)(ws + OFF_Q);    // Q[k]=WA^k*WB, k=1..7 at (k-1)*65536
  uint32_t* PK   = (uint32_t*)(ws + OFF_PK);
  float*    bU   = (float*)(ws + OFF_BU);
  float*    bV   = (float*)(ws + OFF_BV);
  float*    bw   = (float*)(ws + OFF_BW);

  // WA powers (fp32 for accuracy): P2=WA*WA ; Pk=WA*P(k-1)
  k_gemm256<<<256, 256, 0, stream>>>(WA, WA, P);
  for (int k = 3; k <= 8; ++k)
    k_gemm256<<<256, 256, 0, stream>>>(WA, P + (size_t)(k - 3) * 65536,
                                       P + (size_t)(k - 2) * 65536);
  // Q[k] = WA^k * WB
  for (int k = 1; k <= 7; ++k)
    k_gemm256<<<256, 256, 0, stream>>>((k == 1) ? WA : P + (size_t)(k - 2) * 65536, WB,
                                       Q + (size_t)(k - 1) * 65536);
  k_bias<<<1, 256, 0, stream>>>(bA, bB, bC, bD, P, WA, bU, bV, bw);

  // packed bf16 B-frags: [0..7]=WA^k*WB (k=0->WB), [8]=WA, [9]=WA^8, [10]=WC, [11]=WD
  k_pack<<<128, 256, 0, stream>>>(WB, PK);
  for (int k = 1; k <= 7; ++k)
    k_pack<<<128, 256, 0, stream>>>(Q + (size_t)(k - 1) * 65536, PK + (size_t)k * PACK_DW);
  k_pack<<<128, 256, 0, stream>>>(WA, PK + 8ull * PACK_DW);
  k_pack<<<128, 256, 0, stream>>>(P + 6ull * 65536, PK + 9ull * PACK_DW);
  k_pack<<<128, 256, 0, stream>>>(WC, PK + 10ull * PACK_DW);
  k_pack<<<128, 256, 0, stream>>>(WD, PK + 11ull * PACK_DW);

  size_t sm_w = (size_t)(15 * FRAG_DW + PACK_DW) * 4;        // 248 KB (WGP LDS)
  k_wgemm<<<T / 8, 256, sm_w, stream>>>(x, PK, bw, wbuf);

  k_bootstrap<<<1, 512, 0, stream>>>(x, h0, PK, PK + 8ull * PACK_DW, bU, H);

  k_sscan<<<S, 512, 0, stream>>>(wbuf, PK + 9ull * PACK_DW, H, out_h);

  size_t sm_y = (size_t)(4 * FRAG_DW + 2 * PACK_DW) * 4;     // 288 KB
  k_ygemm<<<T / 4, 256, sm_y, stream>>>(x, H, PK + 10ull * PACK_DW, PK + 11ull * PACK_DW,
                                        bV, out_y);
}
